// LSTM_Decoder_Attention_22608707846212
// MI455X (gfx1250) — matrix operand order repaired and checked by reading, not compiled, not run on hardware
//
#include <hip/hip_runtime.h>

// ---------------------------------------------------------------------------
// Types
// ---------------------------------------------------------------------------
typedef __bf16 bf16;
typedef __attribute__((ext_vector_type(16))) __bf16 bf16x16;
typedef __attribute__((ext_vector_type(8)))  __bf16 bf16x8;
typedef __attribute__((ext_vector_type(8)))  float  f32x8;

// Problem constants (from reference)
#define BB   64
#define PP   49
#define IMG  2048
#define EE   512
#define HH   512
#define AA   512
#define VV   10000
#define TT   20
#define GW   2560   // 4*H (i,f,g,o) + H (sentinel)

__device__ __forceinline__ bf16 f2bf(float f) {
    unsigned u = __builtin_bit_cast(unsigned, f);
    unsigned r = u + 0x7FFFu + ((u >> 16) & 1u);   // round-to-nearest-even
    unsigned short s = (unsigned short)(r >> 16);
    return __builtin_bit_cast(bf16, s);
}
__device__ __forceinline__ float sigmoidf(float x) {
    return 1.0f / (1.0f + __expf(-x));
}

// ---------------------------------------------------------------------------
// bf16 WMMA GEMM:  C[M,N](f32) = A[M,K] @ W[N,K]^T (+ bias[N]) (+ C if beta)
// One wave computes a 64x16 strip of C (4 M-tiles sharing one B fragment):
//   - weight (B) fragment fetched once per 4 WMMAs
//   - 4 independent accumulation chains -> no dependent back-to-back WMMA
// Requires M % 64 == 0 (true for all call sites: 64 and 3136).
// Fragment layouts per CDNA5 ISA 7.12.2 (wave32).
// ---------------------------------------------------------------------------
__global__ __launch_bounds__(256) void gemm_bf16_wmma(
    const bf16* __restrict__ A, const bf16* __restrict__ W,
    const float* __restrict__ bias, float* __restrict__ C,
    int M, int N, int K, int ldc, int beta)
{
    const int wave   = threadIdx.x >> 5;
    const int lane   = threadIdx.x & 31;
    const int tilesN = N >> 4;
    const int strip  = blockIdx.x * 8 + wave;            // one 64x16 strip per wave
    if (strip >= (M >> 6) * tilesN) return;              // wave-uniform guard
    const int tm = (strip / tilesN) << 6;
    const int tn = (strip % tilesN) << 4;

    const int half = lane >> 4;                          // 0 or 1
    const int r    = lane & 15;

    // A: lane holds row (tm + mt*16 + r); halfwords j<8 -> k=half*8+j, j>=8 -> +16
    const bf16* aptr = A + (size_t)(tm + r) * K + half * 8;
    const size_t arow = (size_t)16 * K;                  // stride between M-tiles
    // B = W^T: lane holds column (tn+r); halfwords j<8 -> k=half*8+j, j>=8 -> +16
    const bf16* bptr = W + (size_t)(tn + r) * K + half * 8;

    f32x8 acc0 = {}, acc1 = {}, acc2 = {}, acc3 = {};
    union frag { bf16x16 v; bf16x8 h[2]; };
    for (int k = 0; k < K; k += 32) {
        frag bfr, a0, a1, a2, a3;
        bfr.h[0] = *(const bf16x8*)(bptr + k);
        bfr.h[1] = *(const bf16x8*)(bptr + k + 16);
        a0.h[0] = *(const bf16x8*)(aptr + k);
        a0.h[1] = *(const bf16x8*)(aptr + k + 16);
        a1.h[0] = *(const bf16x8*)(aptr + arow + k);
        a1.h[1] = *(const bf16x8*)(aptr + arow + k + 16);
        a2.h[0] = *(const bf16x8*)(aptr + 2 * arow + k);
        a2.h[1] = *(const bf16x8*)(aptr + 2 * arow + k + 16);
        a3.h[0] = *(const bf16x8*)(aptr + 3 * arow + k);
        a3.h[1] = *(const bf16x8*)(aptr + 3 * arow + k + 16);
        acc0 = __builtin_amdgcn_wmma_f32_16x16x32_bf16(false, a0.v, false, bfr.v, (short)0, acc0, false, false);
        acc1 = __builtin_amdgcn_wmma_f32_16x16x32_bf16(false, a1.v, false, bfr.v, (short)0, acc1, false, false);
        acc2 = __builtin_amdgcn_wmma_f32_16x16x32_bf16(false, a2.v, false, bfr.v, (short)0, acc2, false, false);
        acc3 = __builtin_amdgcn_wmma_f32_16x16x32_bf16(false, a3.v, false, bfr.v, (short)0, acc3, false, false);
    }

    const float bv = bias ? bias[tn + r] : 0.0f;
    f32x8 accs[4] = {acc0, acc1, acc2, acc3};
#pragma unroll
    for (int mt = 0; mt < 4; ++mt) {
#pragma unroll
        for (int i = 0; i < 8; ++i) {
            const int m = tm + mt * 16 + i + half * 8;   // D layout: VGPR i -> M = i + half*8
            float* cp = C + (size_t)m * ldc + tn + r;
            float v = accs[mt][i] + bv;
            if (beta) v += *cp;
            *cp = v;
        }
    }
}

// ---------------------------------------------------------------------------
// fp32 -> bf16 conversion
// ---------------------------------------------------------------------------
__global__ void f32_to_bf16_k(const float* __restrict__ src, bf16* __restrict__ dst, int n) {
    int i = blockIdx.x * blockDim.x + threadIdx.x;
    if (i < n) dst[i] = f2bf(src[i]);
}

// bcomb[0:2048] = b_ih + b_hh ; bcomb[2048:2560] = bx + bh
__global__ void build_bcomb(const float* __restrict__ b_ih, const float* __restrict__ b_hh,
                            const float* __restrict__ bx,   const float* __restrict__ bh,
                            float* __restrict__ bcomb) {
    int i = blockIdx.x * blockDim.x + threadIdx.x;
    if (i < 4 * HH)           bcomb[i] = b_ih[i] + b_hh[i];
    else if (i < GW)          bcomb[i] = bx[i - 4 * HH] + bh[i - 4 * HH];
}

// Gather embeddings: out[t,b,:] = bf16(embed[cap[b,t],:])
__global__ void embed_gather(const float* __restrict__ embed, const int* __restrict__ cap,
                             bf16* __restrict__ out) {
    int idx = blockIdx.x * blockDim.x + threadIdx.x;       // TT*BB*EE
    if (idx >= TT * BB * EE) return;
    int e   = idx & (EE - 1);
    int row = idx >> 9;                                    // t*BB + b
    int t = row >> 6, b = row & (BB - 1);
    int tok = cap[b * TT + t];
    out[idx] = f2bf(embed[(size_t)tok * EE + e]);
}

// ---------------------------------------------------------------------------
// LSTM cell + sentinel (uses gates[64,2560] = [i f g o | s-preact])
// ---------------------------------------------------------------------------
__global__ void lstm_cell(const float* __restrict__ gates,
                          float* __restrict__ c_io,
                          float* __restrict__ h_f, bf16* __restrict__ h_bf,
                          float* __restrict__ st_f, bf16* __restrict__ st_bf) {
    int idx = blockIdx.x * blockDim.x + threadIdx.x;       // BB*HH
    if (idx >= BB * HH) return;
    int b = idx >> 9, j = idx & (HH - 1);
    const float* g = gates + (size_t)b * GW;
    float gi = g[j], gf = g[HH + j], gg = g[2 * HH + j], go = g[3 * HH + j], gs = g[4 * HH + j];
    float c  = sigmoidf(gf) * c_io[idx] + sigmoidf(gi) * tanhf(gg);
    float tc = tanhf(c);
    float h  = sigmoidf(go) * tc;
    float s  = sigmoidf(gs) * tc;                          // sentinel (old h already in gs)
    c_io[idx] = c;
    h_f[idx]  = h;  h_bf[idx]  = f2bf(h);
    st_f[idx] = s;  st_bf[idx] = f2bf(s);
}

// ---------------------------------------------------------------------------
// Attention scores + softmax + context + y = h_new + context (one block per b)
// ---------------------------------------------------------------------------
__global__ __launch_bounds__(256) void attention_ctx(
    const float* __restrict__ vis_att,  // [B,P,A]
    const float* __restrict__ attS,     // [B,A]  (sentinel proj)
    const float* __restrict__ attH,     // [B,A]  (hidden proj)
    const float* __restrict__ Watt,     // [A]
    const float* __restrict__ batt,     // [1]
    const float* __restrict__ sp_h,     // [B,P,H]
    const float* __restrict__ st_f,     // [B,H]
    const float* __restrict__ h_f,      // [B,H]
    bf16* __restrict__ y_bf)            // [B,H]
{
    __shared__ float sh_hid[AA];
    __shared__ float sh_w[AA];
    __shared__ float sh_alpha[PP + 1];
    __shared__ float sh_sum;

    const int b   = blockIdx.x;
    const int tid = threadIdx.x;
    for (int i = tid; i < AA; i += 256) {
        sh_hid[i] = attH[(size_t)b * AA + i];
        sh_w[i]   = Watt[i];
    }
    __syncthreads();

    const int wave = tid >> 5, lane = tid & 31;
    for (int p = wave; p < PP + 1; p += 8) {
        const float* row = (p < PP) ? (vis_att + ((size_t)b * PP + p) * AA)
                                    : (attS + (size_t)b * AA);
        float partial = 0.0f;
        for (int a = lane; a < AA; a += 32)
            partial += sh_w[a] * tanhf(row[a] + sh_hid[a]);
        for (int off = 16; off; off >>= 1)
            partial += __shfl_xor(partial, off, 32);
        if (lane == 0) sh_alpha[p] = partial + batt[0];
    }
    __syncthreads();

    if (tid == 0) {
        float mx = sh_alpha[0];
        for (int p = 1; p < PP + 1; ++p) mx = fmaxf(mx, sh_alpha[p]);
        float s = 0.0f;
        for (int p = 0; p < PP + 1; ++p) {
            float e = __expf(sh_alpha[p] - mx);
            sh_alpha[p] = e; s += e;
        }
        sh_sum = s;
    }
    __syncthreads();
    const float inv = 1.0f / sh_sum;

    for (int a = tid; a < HH; a += 256) {
        float ctx = 0.0f;
        for (int p = 0; p < PP; ++p)
            ctx += sh_alpha[p] * sp_h[((size_t)b * PP + p) * HH + a];
        ctx += sh_alpha[PP] * st_f[(size_t)b * HH + a];
        ctx *= inv;
        y_bf[(size_t)b * HH + a] = f2bf(h_f[(size_t)b * HH + a] + ctx);
    }
}

// ---------------------------------------------------------------------------
// Host orchestration
// ---------------------------------------------------------------------------
extern "C" void kernel_launch(void* const* d_in, const int* in_sizes, int n_in,
                              void* d_out, int out_size, void* d_ws, size_t ws_size,
                              hipStream_t stream) {
    // Inputs (setup_inputs dict order)
    const float* gfeat = (const float*)d_in[0];
    const float* sfeat = (const float*)d_in[1];
    const int*   cap   = (const int*)d_in[2];
    const float* embed = (const float*)d_in[4];
    const float* W_ih  = (const float*)d_in[5];
    const float* b_ih  = (const float*)d_in[6];
    const float* W_hh  = (const float*)d_in[7];
    const float* b_hh  = (const float*)d_in[8];
    const float* Wx    = (const float*)d_in[9];
    const float* bx    = (const float*)d_in[10];
    const float* Wh    = (const float*)d_in[11];
    const float* bh    = (const float*)d_in[12];
    const float* Wimg  = (const float*)d_in[13];
    const float* bimg  = (const float*)d_in[14];
    const float* Wsen  = (const float*)d_in[15];
    const float* bsen  = (const float*)d_in[16];
    const float* Wvis  = (const float*)d_in[17];
    const float* bvis  = (const float*)d_in[18];
    const float* Whid  = (const float*)d_in[19];
    const float* bhid  = (const float*)d_in[20];
    const float* Watt  = (const float*)d_in[21];
    const float* batt  = (const float*)d_in[22];
    const float* Winh  = (const float*)d_in[23];
    const float* binh  = (const float*)d_in[24];
    const float* Winc  = (const float*)d_in[25];
    const float* binc  = (const float*)d_in[26];
    const float* Wfc   = (const float*)d_in[27];
    const float* bfc   = (const float*)d_in[28];
    float* out = (float*)d_out;

    // ---- workspace carving ----
    char* ws = (char*)d_ws;
    size_t off = 0;
    auto alloc = [&](size_t bytes) -> void* {
        off = (off + 255) & ~(size_t)255;
        void* p = ws + off;
        off += bytes;
        return p;
    };
    // bf16 buffers
    bf16* w1comb = (bf16*)alloc((size_t)GW * EE * 2);        // [W_ih;Wx]  (acts on e)
    bf16* w2comb = (bf16*)alloc((size_t)GW * HH * 2);        // [W_hh;Wh]  (acts on old h)
    bf16* wimg_b = (bf16*)alloc((size_t)HH * IMG * 2);
    bf16* wsen_b = (bf16*)alloc((size_t)AA * HH * 2);
    bf16* wvis_b = (bf16*)alloc((size_t)AA * HH * 2);
    bf16* whid_b = (bf16*)alloc((size_t)AA * HH * 2);
    bf16* winh_b = (bf16*)alloc((size_t)HH * IMG * 2);
    bf16* winc_b = (bf16*)alloc((size_t)HH * IMG * 2);
    bf16* wfc_b  = (bf16*)alloc((size_t)VV * HH * 2);
    bf16* gf_b   = (bf16*)alloc((size_t)BB * IMG * 2);
    bf16* spf_b  = (bf16*)alloc((size_t)BB * PP * IMG * 2);
    bf16* sph_b  = (bf16*)alloc((size_t)BB * PP * HH * 2);
    bf16* emb_b  = (bf16*)alloc((size_t)TT * BB * EE * 2);   // [T,B,E]
    bf16* h_b    = (bf16*)alloc((size_t)BB * HH * 2);
    bf16* st_b   = (bf16*)alloc((size_t)BB * HH * 2);
    bf16* y_b    = (bf16*)alloc((size_t)BB * HH * 2);
    // f32 buffers
    float* bcomb  = (float*)alloc((size_t)GW * 4);
    float* h_f    = (float*)alloc((size_t)BB * HH * 4);
    float* c_f    = (float*)alloc((size_t)BB * HH * 4);
    float* st_f   = (float*)alloc((size_t)BB * HH * 4);
    float* gates  = (float*)alloc((size_t)BB * GW * 4);
    float* sph_f  = (float*)alloc((size_t)BB * PP * HH * 4);
    float* visatt = (float*)alloc((size_t)BB * PP * AA * 4);
    float* attS   = (float*)alloc((size_t)BB * AA * 4);
    float* attH   = (float*)alloc((size_t)BB * AA * 4);
    (void)ws_size; (void)n_in; (void)in_sizes; (void)out_size;

    auto conv = [&](const float* src, bf16* dst, int n) {
        f32_to_bf16_k<<<(n + 255) / 256, 256, 0, stream>>>(src, dst, n);
    };
    auto gemm = [&](const bf16* A, const bf16* W, const float* bias, float* C,
                    int M, int N, int K, int ldc, int beta) {
        int strips = (M / 64) * (N / 16);                // one 64x16 strip per wave
        int blocks = (strips + 7) / 8;
        gemm_bf16_wmma<<<blocks, 256, 0, stream>>>(A, W, bias, C, M, N, K, ldc, beta);
    };

    // ---- one-time per-call conversions (deterministic) ----
    conv(W_ih, w1comb,                 4 * HH * EE);
    conv(Wx,   w1comb + 4 * HH * EE,   HH * EE);
    conv(W_hh, w2comb,                 4 * HH * HH);
    conv(Wh,   w2comb + 4 * HH * HH,   HH * HH);
    conv(Wimg, wimg_b, HH * IMG);
    conv(Wsen, wsen_b, AA * HH);
    conv(Wvis, wvis_b, AA * HH);
    conv(Whid, whid_b, AA * HH);
    conv(Winh, winh_b, HH * IMG);
    conv(Winc, winc_b, HH * IMG);
    conv(Wfc,  wfc_b,  VV * HH);
    conv(gfeat, gf_b,  BB * IMG);
    conv(sfeat, spf_b, BB * PP * IMG);
    build_bcomb<<<(GW + 255) / 256, 256, 0, stream>>>(b_ih, b_hh, bx, bh, bcomb);
    embed_gather<<<(TT * BB * EE + 255) / 256, 256, 0, stream>>>(embed, cap, emb_b);

    // ---- precompute ----
    gemm(gf_b,  winh_b, binh, h_f,    BB, HH, IMG, HH, 0);              // h0
    gemm(gf_b,  winc_b, binc, c_f,    BB, HH, IMG, HH, 0);              // c0
    conv(h_f, h_b, BB * HH);
    gemm(spf_b, wimg_b, bimg, sph_f,  BB * PP, HH, IMG, HH, 0);         // sp_h
    conv(sph_f, sph_b, BB * PP * HH);
    gemm(sph_b, wvis_b, bvis, visatt, BB * PP, AA, HH, AA, 0);          // vis_att

    // ---- T sequential decode steps ----
    for (int t = 0; t < TT; ++t) {
        const bf16* e_t = emb_b + (size_t)t * BB * EE;
        // fused gates + sentinel preact: [64,2560] = e@[W_ih;Wx]^T + h_old@[W_hh;Wh]^T + b
        gemm(e_t, w1comb, bcomb, gates, BB, GW, EE, GW, 0);
        gemm(h_b, w2comb, nullptr, gates, BB, GW, HH, GW, 1);
        lstm_cell<<<(BB * HH + 255) / 256, 256, 0, stream>>>(
            gates, c_f, h_f, h_b, st_f, st_b);
        gemm(st_b, wsen_b, bsen, attS, BB, AA, HH, AA, 0);              // sentinel proj
        gemm(h_b,  whid_b, bhid, attH, BB, AA, HH, AA, 0);              // hidden proj
        attention_ctx<<<BB, 256, 0, stream>>>(
            visatt, attS, attH, Watt, batt, sph_f, st_f, h_f, y_b);
        // pred -> out[b, t, :]   (ldc = T*V so rows land at b*T*V + t*V)
        gemm(y_b, wfc_b, bfc, out + (size_t)t * VV, BB, VV, HH, TT * VV, 0);
    }
}
